// HullsCMPNN_2774548873289
// MI455X (gfx1250) — compile-verified, matrix operand order checked by reading, not run on hardware
//
#include <hip/hip_runtime.h>

// ============================================================================
// CEGNN (Clifford Cl(5,0) GNN) forward on MI455X / gfx1250.
// Strategy: fuse each full CEMLP (2 blocks) per workgroup in LDS (~287KB of
// the 320KB/WGP), run all channel-mixing mv_linears as v_wmma_f32_16x16x32_bf16
// (K=32 == channel count -> one WMMA per 16x16 tile), keep the 1024-term
// Cayley geometric product as fully-unrolled VALU FMAs with constexpr tables.
// Round 2: branch-free bias (v_cndmask), b128 DS fragment loads, gp_w in LDS.
// ============================================================================

typedef __attribute__((ext_vector_type(16))) __bf16 bf16x16;
typedef __attribute__((ext_vector_type(8)))  float  f32x8;

#define DEVFN __device__ __forceinline__

constexpr int NBL   = 32;   // blades
constexpr int CH    = 32;   // channels (all block outputs)
constexpr int TIL   = 16;   // edges/nodes per workgroup tile (== WMMA M)
constexpr int NTHR  = 512;  // TIL * CH threads -> 16 waves
constexpr int NPATH = 56;   // realized grade-triples in Cl(5,0)

// ---------------- compile-time Cayley tables -------------------------------
struct Tab { int blade[32]; int grade[32]; int KS[32][32]; int PS[32][32]; float TS[32][32]; };

constexpr int popc5(int v){ int c=0; for(int i=0;i<6;++i) c += (v>>i)&1; return c; }
constexpr bool realized(int x,int y,int z){
  if(((x+z-y)&1)!=0) return false;
  int lo = x>z ? x-z : z-x;
  int hi = (x+z) < (10-x-z) ? (x+z) : (10-x-z);
  return y>=lo && y<=hi;
}
constexpr int path_id(int x,int y,int z){
  int id=0;
  for(int a=0;a<6;++a) for(int b=0;b<6;++b) for(int c=0;c<6;++c){
    if(a==x && b==y && c==z) return id;
    if(realized(a,b,c)) ++id;
  }
  return 0;
}
constexpr Tab make_tab(){
  Tab t{};
  int pos[32] = {};
  int idx=0;
  for(int g=0; g<6; ++g)
    for(int b=0; b<32; ++b)
      if(popc5(b)==g){ t.blade[idx]=b; pos[b]=idx; t.grade[idx]=g; ++idx; }
  for(int j=0;j<32;++j)
    for(int q=0;q<32;++q){
      int a  = t.blade[q];
      int bb = a ^ t.blade[j];
      int s=0, aa=a>>1;
      while(aa){ s += popc5(aa & bb); aa >>= 1; }
      t.KS[j][q] = pos[bb];
      t.TS[j][q] = (s&1) ? -1.f : 1.f;
      t.PS[j][q] = path_id(popc5(a), popc5(a^bb), popc5(bb));
    }
  return t;
}
static constexpr Tab kT = make_tab();

// ---------------- helpers ---------------------------------------------------
DEVFN __bf16 tobf(float f){
  union { float f; unsigned u; } a; a.f = f;
  unsigned r = (a.u + 0x7FFFu + ((a.u >> 16) & 1u)) >> 16;   // RNE
  union { unsigned short s; __bf16 b; } o; o.s = (unsigned short)r;
  return o.b;
}

struct BlockParams {   // leaves in alphabetical order of reference dict keys
  const float *gp_w, *left_b, *left_w, *lin_b, *lin_w,
              *ln_a, *norm_a, *right_w, *silu_a, *silu_b;
};

// LDS pool layout (bytes)
constexpr unsigned LDS_A   = 0;        // 128 KB : X / (XA2 + R)
constexpr unsigned LDS_B   = 131072;   //  64 KB : Y
constexpr unsigned LDS_C   = 196608;   //  64 KB : L / block output / XAlin
constexpr unsigned LDS_W   = 262144;   //  24 KB : bf16 weights [6][32][Cin]
constexpr unsigned LDS_IDX = 286720;   // 128 B  : edge indices
constexpr unsigned LDS_G   = 286848;   // 7 KB   : gp_w tile [32][56]
constexpr unsigned LDS_TOTAL = 294016;

// stage bf16 A-operand copy: dst[t][i][m] = bf16(src[t][m][i])
template<int CIN>
DEVFN void stage_x(const float* __restrict__ src, __bf16* __restrict__ dst, int tid){
  const int total = TIL * NBL * CIN;
  for (int e = tid; e < total; e += NTHR){
    const int m = e % CIN, i = (e / CIN) % NBL, t = e / (CIN * NBL);
    dst[e] = tobf(src[(t*CIN + m)*NBL + i]);
  }
}
// stage bf16 weights: dst[g][n][m] = bf16(w[n][m][g]);  w global (32,CIN,6)
template<int CIN>
DEVFN void stage_w(const float* __restrict__ w, __bf16* __restrict__ dst, int tid){
  const int total = 6 * 32 * CIN;
  for (int e = tid; e < total; e += NTHR){
    const int m = e % CIN, n = (e / CIN) % 32, g = e / (32 * CIN);
    dst[e] = tobf(w[(n*CIN + m)*6 + g]);
  }
}

// ---------------- WMMA mv_linear -------------------------------------------
// out[t][n][i] = sum_m X[t][m][i] * W[n][m][grade(i)]  (+ bias[n] on blade 0)
// XA: LDS bf16 [TIL][NBL][CIN]   WB: LDS bf16 [6][32][CIN]   dst: [TIL][32][NBL] f32
template<int CIN, bool HB>
DEVFN void mvlin_wmma(const __bf16* __restrict__ XA, const __bf16* __restrict__ WB,
                      float* __restrict__ dst, const float* __restrict__ bias, int tid)
{
  const int wave = tid >> 5, lane = tid & 31;
  const int ln = lane & 15, hi = lane >> 4;
  #pragma unroll
  for (int r = 0; r < 4; ++r){                       // 64 tiles / 16 waves
    const int tile = wave + (r << 4);
    const int i  = tile >> 1;                        // blade
    const int n0 = (tile & 1) << 4;                  // out-channel tile
    const int g  = kT.grade[i];
    const float bv = HB ? bias[n0 + ln] : 0.f;       // unconditional, once/tile
    f32x8 acc = {0.f,0.f,0.f,0.f,0.f,0.f,0.f,0.f};
    #pragma unroll
    for (int kk = 0; kk < CIN; kk += 32){
      // 16-bit A layout (ISA 7.12.2): lane's 16 halves are dwords
      // [hi*4 .. hi*4+3] and [8+hi*4 .. 8+hi*4+3]  -> two ds_load_b128 each
      union { uint4 q4[2]; bf16x16 v; } Af, Bf;
      const uint4* ap = (const uint4*)&XA[(ln*NBL + i)*CIN + kk];
      const uint4* bp = (const uint4*)&WB[((g<<5) + n0 + ln)*CIN + kk];
      Af.q4[0] = ap[hi];     Af.q4[1] = ap[2 + hi];
      Bf.q4[0] = bp[hi];     Bf.q4[1] = bp[2 + hi];
      acc = __builtin_amdgcn_wmma_f32_16x16x32_bf16(false, Af.v, false, Bf.v,
                                                    (short)0, acc, false, false);
    }
    #pragma unroll
    for (int q = 0; q < 8; ++q){
      const int t = q + (hi << 3);                   // D row = vreg + 8*hi
      float v = acc[q];
      if (HB) v += (i == 0) ? bv : 0.f;              // v_cndmask, no branches
      dst[(t*CH + n0 + ln)*NBL + i] = v;
    }
  }
}

// ---------------- elementwise pieces (thread == one (t,c) multivector) -----
DEVFN void silu_ip(float* __restrict__ Y, const float* __restrict__ a6,
                   const float* __restrict__ b6, int tid){
  const int t = tid >> 5, c = tid & 31;
  float* x = &Y[(t*CH + c)*NBL];
  float xr[32];
  #pragma unroll
  for (int i = 0; i < 32; ++i) xr[i] = x[i];
  float q[6] = {0,0,0,0,0,0};
  #pragma unroll
  for (int i = 0; i < 32; ++i) q[kT.grade[i]] += xr[i]*xr[i];
  float s[6];
  #pragma unroll
  for (int g = 0; g < 6; ++g){
    const float inv = (g==0) ? xr[0] : q[g];
    const float z   = a6[c*6+g]*inv + b6[c*6+g];
    s[g] = 1.f / (1.f + __expf(-z));
  }
  #pragma unroll
  for (int i = 0; i < 32; ++i) x[i] = xr[i] * s[kT.grade[i]];
}

DEVFN void gnorm_ip(float* __restrict__ Y, const float* __restrict__ a6, int tid){
  const int t = tid >> 5, c = tid & 31;
  float* x = &Y[(t*CH + c)*NBL];
  float xr[32];
  #pragma unroll
  for (int i = 0; i < 32; ++i) xr[i] = x[i];
  float q[6] = {0,0,0,0,0,0};
  #pragma unroll
  for (int i = 0; i < 32; ++i) q[kT.grade[i]] += xr[i]*xr[i];
  float sc[6];
  #pragma unroll
  for (int g = 0; g < 6; ++g){
    const float n  = __fsqrt_rn(q[g] + 1e-12f);
    const float sg = 1.f / (1.f + __expf(-a6[c*6+g]));
    const float nn = sg * (n - 1.f) + 1.f;
    sc[g] = 1.f / (nn + 1e-6f);
  }
  #pragma unroll
  for (int i = 0; i < 32; ++i) x[i] = xr[i] * sc[kT.grade[i]];
}

// out = (left + geometric_product(x=Y, y=R; gp_w)) / sqrt(2), in-place on Ob
DEVFN void gp_out(const float* __restrict__ Yb, const float* __restrict__ Rb,
                  float* __restrict__ Ob, const float* __restrict__ GW, int tid){
  const int t = tid >> 5, c = tid & 31;
  const float* x = &Yb[(t*CH + c)*NBL];
  const float* y = &Rb[(t*CH + c)*NBL];
  float*       o = &Ob[(t*CH + c)*NBL];
  float xr[32], yr[32], wr[NPATH];
  #pragma unroll
  for (int i = 0; i < 32; ++i){ xr[i] = x[i]; yr[i] = y[i]; }
  #pragma unroll
  for (int p = 0; p < NPATH; ++p) wr[p] = GW[c*NPATH + p];
  const float inv_s2 = 0.70710678118654752f;
  #pragma unroll
  for (int j = 0; j < 32; ++j){
    float acc = 0.f;
    #pragma unroll
    for (int q = 0; q < 32; ++q)                       // indices constant-fold
      acc += kT.TS[j][q] * xr[q] * yr[kT.KS[j][q]] * wr[kT.PS[j][q]];
    o[j] = (o[j] + acc) * inv_s2;
  }
}

DEVFN void mvln_ip(float* __restrict__ Y, const float* __restrict__ lna, int tid){
  const int t = tid >> 5, c = tid & 31;                 // 32 channels == one wave
  float* x = &Y[(t*CH + c)*NBL];
  float ss = 0.f;
  #pragma unroll
  for (int i = 0; i < 32; ++i) ss += x[i]*x[i];
  float tot = __fsqrt_rn(ss + 1e-12f);
  #pragma unroll
  for (int off = 16; off > 0; off >>= 1) tot += __shfl_xor(tot, off, 32);
  const float sc = lna[c] / (tot * (1.f/32.f) + 1e-6f);
  #pragma unroll
  for (int i = 0; i < 32; ++i) x[i] *= sc;
}

// ---------------- one CEMLP block ------------------------------------------
// in:[TIL][CIN][NBL] f32 LDS.  Output lands in Cf:[TIL][32][NBL].
template<int CIN>
DEVFN void cemlp_block(const float* in, __bf16* XAlin, float* A, float* Bf, float* Cf,
                       __bf16* W, float* GW, const BlockParams P, int tid)
{
  __bf16* XA2 = (__bf16*)A;        // 32 KB, lives where dead X was
  float*  R   = A + 16384;         // 64 KB offset into region A
  __builtin_prefetch(P.right_w, 0, 1);   // global_prefetch_b8
  __builtin_prefetch(P.left_w,  0, 1);
  __builtin_prefetch(P.gp_w,    0, 1);

  stage_w<CIN>(P.lin_w, W, tid);
  stage_x<CIN>(in, XAlin, tid);
  for (int e = tid; e < 32*NPATH; e += NTHR) GW[e] = P.gp_w[e];   // gp_w tile
  __syncthreads();
  mvlin_wmma<CIN, true>(XAlin, W, Bf, P.lin_b, tid);   // Y = lin(X)+b
  __syncthreads();
  silu_ip(Bf, P.silu_a, P.silu_b, tid);                // Y = MVSiLU(Y)
  __syncthreads();
  stage_w<CH>(P.right_w, W, tid);
  stage_x<CH>(Bf, XA2, tid);
  __syncthreads();
  mvlin_wmma<CH, false>(XA2, W, R, nullptr, tid);      // R = right(Y)
  __syncthreads();
  gnorm_ip(R, P.norm_a, tid);                          // R = grade_norm(R)
  stage_w<CH>(P.left_w, W, tid);
  __syncthreads();
  mvlin_wmma<CH, true>(XA2, W, Cf, P.left_b, tid);     // L = left(Y)+b
  __syncthreads();
  gp_out(Bf, R, Cf, GW, tid);                          // (L + gp(Y,R))/sqrt2
  __syncthreads();
  mvln_ip(Cf, P.ln_a, tid);                            // MVLayerNorm
  __syncthreads();
}

// ---------------- kernels ---------------------------------------------------
__global__ void embed_kernel(const float* __restrict__ x, const float* __restrict__ emb_w,
                             const float* __restrict__ emb_b, float* __restrict__ h)
{
  __shared__ float red[5][128];
  const int tid = threadIdx.x;
  const int n   = blockIdx.x * 128 + tid;
  float v[5];
  #pragma unroll
  for (int d = 0; d < 5; ++d){ v[d] = x[n*5 + d]; red[d][tid] = v[d]; }
  __syncthreads();
  for (int s = 64; s > 0; s >>= 1){
    if (tid < s)
      for (int d = 0; d < 5; ++d) red[d][tid] += red[d][tid + s];
    __syncthreads();
  }
  #pragma unroll
  for (int d = 0; d < 5; ++d) v[d] -= red[d][0] * (1.f/128.f);   // center per-graph
  for (int c = 0; c < 32; ++c){
    float* o = &h[(n*CH + c)*NBL];
    const float w = emb_w[c];
    o[0] = emb_b[c];
    #pragma unroll
    for (int d = 0; d < 5; ++d) o[1 + d] = w * v[d];             // grade-1 blades
    #pragma unroll
    for (int i = 6; i < 32; ++i) o[i] = 0.f;
  }
}

__global__ void zero_kernel(float* p, int n){
  const int i = blockIdx.x * blockDim.x + threadIdx.x;
  if (i < n) p[i] = 0.f;
}

__global__ __launch_bounds__(NTHR, 1)
void edge_cemlp_kernel(const float* __restrict__ h, const int* __restrict__ ei,
                       BlockParams p0, BlockParams p1,
                       float* __restrict__ agg, float* __restrict__ cnt, int E)
{
  extern __shared__ char smem[];
  float*  A  = (float*)(smem + LDS_A);
  float*  Bf = (float*)(smem + LDS_B);
  float*  Cf = (float*)(smem + LDS_C);
  __bf16* W  = (__bf16*)(smem + LDS_W);
  int*    ix = (int*)(smem + LDS_IDX);
  float*  GW = (float*)(smem + LDS_G);
  const int tid = threadIdx.x;
  const int e0  = blockIdx.x * TIL;
  if (tid < 2*TIL)
    ix[tid] = (tid < TIL) ? ei[E + e0 + tid]          // dst
                          : ei[e0 + tid - TIL];       // src
  __syncthreads();
  const int t = tid >> 5, c = tid & 31;
  {                                                   // X = h[dst] - h[src]
    const int d = ix[t], s = ix[TIL + t];
    const float* hd = &h[((long)d*CH + c)*NBL];
    const float* hs = &h[((long)s*CH + c)*NBL];
    float* xo = &A[(t*CH + c)*NBL];
    #pragma unroll
    for (int i = 0; i < 32; ++i) xo[i] = hd[i] - hs[i];
  }
  __syncthreads();
  cemlp_block<32>(A,  (__bf16*)Cf, A, Bf, Cf, W, GW, p0, tid);
  cemlp_block<32>(Cf, (__bf16*)A,  A, Bf, Cf, W, GW, p1, tid);
  const int d = ix[t];                                // scatter-sum numerator
  float* ap = &agg[((long)d*CH + c)*NBL];
  const float* sp = &Cf[(t*CH + c)*NBL];
  #pragma unroll
  for (int i = 0; i < 32; ++i) atomicAdd(&ap[i], sp[i]);
  if (c == 0) atomicAdd(&cnt[d], 1.0f);
}

__global__ __launch_bounds__(NTHR, 1)
void node_cemlp_kernel(float* __restrict__ h, const float* __restrict__ agg,
                       const float* __restrict__ cnt,
                       BlockParams p0, BlockParams p1)
{
  extern __shared__ char smem[];
  float*  A  = (float*)(smem + LDS_A);
  float*  Bf = (float*)(smem + LDS_B);
  float*  Cf = (float*)(smem + LDS_C);
  __bf16* W  = (__bf16*)(smem + LDS_W);
  float*  GW = (float*)(smem + LDS_G);
  const int tid = threadIdx.x;
  const int n0  = blockIdx.x * TIL;
  for (int s = tid; s < TIL*64; s += NTHR){           // X = concat(h, agg/cnt)
    const int t = s >> 6, c64 = s & 63;
    const int node = n0 + t;
    float* xo = &A[(t*64 + c64)*NBL];
    if (c64 < 32){
      const float* hp = &h[((long)node*CH + c64)*NBL];
      #pragma unroll
      for (int i = 0; i < 32; ++i) xo[i] = hp[i];
    } else {
      const float icv = 1.f / fmaxf(cnt[node], 1.f);
      const float* ap = &agg[((long)node*CH + (c64 - 32))*NBL];
      #pragma unroll
      for (int i = 0; i < 32; ++i) xo[i] = ap[i] * icv;
    }
  }
  __syncthreads();
  cemlp_block<64>(A,  (__bf16*)Cf, A, Bf, Cf, W, GW, p0, tid);
  cemlp_block<32>(Cf, (__bf16*)A,  A, Bf, Cf, W, GW, p1, tid);
  const int t = tid >> 5, c = tid & 31;               // residual
  const int node = n0 + t;
  float* hp = &h[((long)node*CH + c)*NBL];
  const float* sp = &Cf[(t*CH + c)*NBL];
  #pragma unroll
  for (int i = 0; i < 32; ++i) hp[i] += sp[i];
}

__global__ void readout_kernel(const float* __restrict__ h, const float* __restrict__ proj_w,
                               const float* __restrict__ proj_b, const float* __restrict__ target,
                               float* __restrict__ out)
{
  __shared__ float red[128];
  const int tid = threadIdx.x;
  const int n   = blockIdx.x * 128 + tid;
  float s = proj_b[0];
  for (int c = 0; c < 32; ++c)                        // scalar blade only (g=0)
    s += h[((long)n*CH + c)*NBL] * proj_w[c*6];
  red[tid] = s; __syncthreads();
  for (int q = 64; q > 0; q >>= 1){
    if (tid < q) red[tid] += red[tid + q];
    __syncthreads();
  }
  if (tid == 0){
    const float pred = red[0] * (1.f/128.f);
    const float d = pred - target[blockIdx.x];
    out[blockIdx.x] = d * d;
  }
}

// ---------------- host orchestration ---------------------------------------
extern "C" void kernel_launch(void* const* d_in, const int* in_sizes, int n_in,
                              void* d_out, int out_size, void* d_ws, size_t ws_size,
                              hipStream_t stream)
{
  // jax pytree flatten order (dict keys alphabetical, lists in order):
  // 0: edge_index | 1: emb_b | 2: emb_w | 3..122: 3 layers x (edge blk0, edge blk1,
  // node blk0, node blk1) x {gp_w,left_b,left_w,lin_b,lin_w,ln_a,norm_a,right_w,
  // silu_a,silu_b} | 123: proj_b | 124: proj_w | 125: target | 126: x
  auto F = [&](int i){ return (const float*)d_in[i]; };
  const int* ei = (const int*)d_in[0];
  int pi = 1;
  const float* emb_b = F(pi++); const float* emb_w = F(pi++);
  BlockParams bp[12];
  for (int k = 0; k < 12; ++k){
    BlockParams b;
    b.gp_w = F(pi+0); b.left_b = F(pi+1); b.left_w  = F(pi+2); b.lin_b  = F(pi+3);
    b.lin_w= F(pi+4); b.ln_a   = F(pi+5); b.norm_a  = F(pi+6); b.right_w= F(pi+7);
    b.silu_a = F(pi+8); b.silu_b = F(pi+9);
    pi += 10; bp[k] = b;
  }
  const float* proj_b = F(pi++); const float* proj_w = F(pi++);
  const float* target = F(pi++); const float* x      = F(pi++);

  const int E = in_sizes[0] / 2;     // 32768
  const int N = 4096;
  float* h   = (float*)d_ws;         // 16 MB
  float* agg = h   + (size_t)N*CH*NBL;   // 16 MB
  float* cnt = agg + (size_t)N*CH*NBL;   // 16 KB   (agg & cnt contiguous)

  hipFuncSetAttribute((const void*)edge_cemlp_kernel,
                      hipFuncAttributeMaxDynamicSharedMemorySize, LDS_TOTAL);
  hipFuncSetAttribute((const void*)node_cemlp_kernel,
                      hipFuncAttributeMaxDynamicSharedMemorySize, LDS_TOTAL);

  embed_kernel<<<N/128, 128, 0, stream>>>(x, emb_w, emb_b, h);
  for (int L = 0; L < 3; ++L){
    const int nz = N*CH*NBL + N;
    zero_kernel<<<(nz + 255)/256, 256, 0, stream>>>(agg, nz);
    edge_cemlp_kernel<<<E/TIL, NTHR, LDS_TOTAL, stream>>>(h, ei, bp[4*L+0], bp[4*L+1],
                                                          agg, cnt, E);
    node_cemlp_kernel<<<N/TIL, NTHR, LDS_TOTAL, stream>>>(h, agg, cnt,
                                                          bp[4*L+2], bp[4*L+3]);
  }
  readout_kernel<<<32, 128, 0, stream>>>(h, proj_w, proj_b, target, (float*)d_out);
}
